// OT_AutoRec_53446573031855
// MI455X (gfx1250) — matrix operand order repaired and checked by reading, not compiled, not run on hardware
//
#include <hip/hip_runtime.h>

// ---------------------------------------------------------------------------
// OT_AutoRec for MI455X (gfx1250, wave32, WMMA)
//   embedding = sigmoid(x @ W_enc^T + b_enc)           (2048 x 800)
//   pred      = embedding @ W_dec^T + b_dec            (2048 x 20000)
//   cost      = 1 - normalize(E) @ normalize(Eo)^T     (2048 x 2048)
//   IPOT(50 iters) -> ot_loss = sum(C * T)
// GEMMs: f32 -> f16 LDS staging, v_wmma_f32_16x16x32_f16, f32 accumulate,
// double-buffered LDS pipeline, 2x2 accumulator blocking per wave.
// ---------------------------------------------------------------------------

typedef __attribute__((ext_vector_type(16))) _Float16 v16h;
typedef __attribute__((ext_vector_type(8)))  _Float16 v8h;
typedef __attribute__((ext_vector_type(8)))  float    v8f;

#define BM 128
#define BN 64
#define BK 32
#define LDSH 40   // halves per LDS row (80B: 16B-aligned rows, conflict-free)

struct TileRegs {
  float4 a[2][2];   // 2 segments x 8 consecutive floats of the A tile
  float4 b[2];      // 1 segment  x 8 consecutive floats of the B tile
};

// Load one 128x32 A tile segment pair + one 64x32 B tile segment into regs.
// A is always in-bounds (M%128==0, K%32==0); B rows clamped + zeroed (no exec
// branch).
__device__ __forceinline__ void load_tile(const float* __restrict__ A,
                                          const float* __restrict__ B,
                                          int bm, int bn, int N, int K, int k0,
                                          int tid, TileRegs& r) {
#pragma unroll
  for (int it = 0; it < 2; ++it) {
    const int seg = tid + it * 256;     // 0..511 -> 128 rows x 4 segments
    const int row = seg >> 2;
    const int kc  = (seg & 3) << 3;
    const float* p = A + (size_t)(bm + row) * K + k0 + kc;
    r.a[it][0] = *reinterpret_cast<const float4*>(p);
    r.a[it][1] = *reinterpret_cast<const float4*>(p + 4);
    __builtin_prefetch(p + BK, 0, 3);   // warm next K tile (gfx1250 prefetch)
  }
  {
    const int row  = tid >> 2;          // 0..63
    const int kc   = (tid & 3) << 3;
    const int nrow = bn + row;
    const int cr   = nrow < N ? nrow : N - 1;   // clamp: branchless edge
    const float* p = B + (size_t)cr * K + k0 + kc;
    float4 b0 = *reinterpret_cast<const float4*>(p);
    float4 b1 = *reinterpret_cast<const float4*>(p + 4);
    if (nrow >= N) {                    // v_cndmask, not exec branch
      b0 = make_float4(0.f, 0.f, 0.f, 0.f);
      b1 = make_float4(0.f, 0.f, 0.f, 0.f);
    }
    r.b[0] = b0;
    r.b[1] = b1;
    __builtin_prefetch(p + BK, 0, 3);
  }
}

__device__ __forceinline__ void cvt_store8(_Float16* d, const float4& x,
                                           const float4& y) {
  v8h h;
  h[0] = (_Float16)x.x; h[1] = (_Float16)x.y;
  h[2] = (_Float16)x.z; h[3] = (_Float16)x.w;
  h[4] = (_Float16)y.x; h[5] = (_Float16)y.y;
  h[6] = (_Float16)y.z; h[7] = (_Float16)y.w;
  *reinterpret_cast<v8h*>(d) = h;       // one ds_store_b128
}

__device__ __forceinline__ void store_tile(_Float16* As, _Float16* Bs, int tid,
                                           const TileRegs& r) {
#pragma unroll
  for (int it = 0; it < 2; ++it) {
    const int seg = tid + it * 256;
    const int row = seg >> 2;
    const int kc  = (seg & 3) << 3;
    cvt_store8(&As[row * LDSH + kc], r.a[it][0], r.a[it][1]);
  }
  const int row = tid >> 2;
  const int kc  = (tid & 3) << 3;
  cvt_store8(&Bs[row * LDSH + kc], r.b[0], r.b[1]);
}

// Fragments per documented CDNA5 wave32 layouts.
// A 16x32 f16: lane<16 -> K {0..7,16..23}; lane>=16 -> K {8..15,24..31}.
// B 32x16 f16: lane<16 -> K {0..7,16..23}; lane>=16 -> K {8..15,24..31}.
__device__ __forceinline__ void compute_step(const _Float16* As,
                                             const _Float16* Bs, int wm, int wn,
                                             int hi, int lr, v8f& c00, v8f& c01,
                                             v8f& c10, v8f& c11) {
  union U { v16h v; v8h h[2]; };
  U a0, a1, b0, b1;
  const _Float16* ar0 = As + (wm + lr) * LDSH;
  const _Float16* ar1 = As + (wm + 16 + lr) * LDSH;
  a0.h[0] = *reinterpret_cast<const v8h*>(ar0 + hi * 8);
  a0.h[1] = *reinterpret_cast<const v8h*>(ar0 + 16 + hi * 8);
  a1.h[0] = *reinterpret_cast<const v8h*>(ar1 + hi * 8);
  a1.h[1] = *reinterpret_cast<const v8h*>(ar1 + 16 + hi * 8);
  const _Float16* br0 = Bs + (wn + lr) * LDSH + hi * 8;
  const _Float16* br1 = Bs + (wn + 16 + lr) * LDSH + hi * 8;
  b0.h[0] = *reinterpret_cast<const v8h*>(br0);
  b0.h[1] = *reinterpret_cast<const v8h*>(br0 + 16);
  b1.h[0] = *reinterpret_cast<const v8h*>(br1);
  b1.h[1] = *reinterpret_cast<const v8h*>(br1 + 16);
  c00 = __builtin_amdgcn_wmma_f32_16x16x32_f16(false, a0.v, false, b0.v,
                                               (short)0, c00, false, false);
  c01 = __builtin_amdgcn_wmma_f32_16x16x32_f16(false, a0.v, false, b1.v,
                                               (short)0, c01, false, false);
  c10 = __builtin_amdgcn_wmma_f32_16x16x32_f16(false, a1.v, false, b0.v,
                                               (short)0, c10, false, false);
  c11 = __builtin_amdgcn_wmma_f32_16x16x32_f16(false, a1.v, false, b1.v,
                                               (short)0, c11, false, false);
}

// NT GEMM: out[m][n] = epilogue( sum_k A[m][k] * B[n][k] )
// A: (M,K) row-major f32, B: (N,K) row-major f32.  M%128==0, K%32==0.
// EPI 0: sigmoid(acc + bias[n])  EPI 1: acc + bias[n]
// EPI 2: c = 1-acc -> out; out2 = exp(-2c)   (A-matrix for IPOT, BETA=0.5)
template <int EPI>
__global__ __launch_bounds__(256) void gemm_nt(const float* __restrict__ A,
                                               const float* __restrict__ B,
                                               const float* __restrict__ bias,
                                               float* __restrict__ out,
                                               float* __restrict__ out2,
                                               int M, int N, int K) {
  __shared__ _Float16 As[2][BM * LDSH];
  __shared__ _Float16 Bs[2][BN * LDSH];

  const int tid  = threadIdx.x;
  const int bm   = blockIdx.y * BM;
  const int bn   = blockIdx.x * BN;
  const int wave = tid >> 5;
  const int lane = tid & 31;
  const int wm   = (wave & 3) * 32;   // wave row offset in tile (32 rows)
  const int wn   = (wave >> 2) * 32;  // wave col offset in tile (32 cols)
  const int hi   = lane >> 4;
  const int lr   = lane & 15;

  v8f c00 = {}, c01 = {}, c10 = {}, c11 = {};

  TileRegs r;
  load_tile(A, B, bm, bn, N, K, 0, tid, r);
  store_tile(As[0], Bs[0], tid, r);
  __syncthreads();

  int buf = 0;
  for (int k0 = 0; k0 < K; k0 += BK) {
    const bool more = (k0 + BK) < K;
    if (more) load_tile(A, B, bm, bn, N, K, k0 + BK, tid, r);  // async-ish
    compute_step(As[buf], Bs[buf], wm, wn, hi, lr, c00, c01, c10, c11);
    if (more) {
      store_tile(As[buf ^ 1], Bs[buf ^ 1], tid, r);  // waits loads here
      __syncthreads();
    }
    buf ^= 1;
  }

  // epilogue: C VGPR v -> row (wm + sm*16 + hi*8 + v), col (wn + sn*16 + lr)
#pragma unroll
  for (int sm = 0; sm < 2; ++sm) {
#pragma unroll
    for (int sn = 0; sn < 2; ++sn) {
      v8f& c = sm ? (sn ? c11 : c10) : (sn ? c01 : c00);
      const int n = bn + wn + sn * 16 + lr;
      if (n >= N) continue;
#pragma unroll
      for (int v = 0; v < 8; ++v) {
        const int m = bm + wm + sm * 16 + hi * 8 + v;
        const float acc = c[v];
        float val;
        if (EPI == 0) {
          val = acc + bias[n];
          val = 1.0f / (1.0f + expf(-val));
        } else if (EPI == 1) {
          val = acc + bias[n];
        } else {
          val = 1.0f - acc;
        }
        out[(size_t)m * N + n] = val;
        if (EPI == 2) out2[(size_t)m * N + n] = expf(-2.0f * val);
      }
    }
  }
}

// ---------------------------------------------------------------------------

__global__ __launch_bounds__(256) void row_normalize(const float* __restrict__ in,
                                                     float* __restrict__ out,
                                                     int F) {
  __shared__ float sm[256];
  const int row = blockIdx.x;
  const float* r = in + (size_t)row * F;
  float s = 0.f;
  for (int j = threadIdx.x; j < F; j += 256) { const float v = r[j]; s += v * v; }
  sm[threadIdx.x] = s;
  __syncthreads();
  for (int ofs = 128; ofs > 0; ofs >>= 1) {
    if (threadIdx.x < ofs) sm[threadIdx.x] += sm[threadIdx.x + ofs];
    __syncthreads();
  }
  const float inv = 1.0f / (sqrtf(sm[0]) + 1e-8f);
  for (int j = threadIdx.x; j < F; j += 256) out[(size_t)row * F + j] = r[j] * inv;
}

__global__ void fill_f32(float* p, size_t n, float v) {
  for (size_t i = (size_t)blockIdx.x * 256 + threadIdx.x; i < n;
       i += (size_t)gridDim.x * 256)
    p[i] = v;
}

// delta[i] = (1/n) / sum_j A[i,j]*T[i,j]*sigma[j]
__global__ __launch_bounds__(256) void ipot_delta(const float* __restrict__ A,
                                                  const float* __restrict__ T,
                                                  const float* __restrict__ sigma,
                                                  float* __restrict__ delta, int n) {
  __shared__ float sm[256];
  const int i = blockIdx.x;
  const float* ar = A + (size_t)i * n;
  const float* tr = T + (size_t)i * n;
  float s = 0.f;
  for (int j = threadIdx.x; j < n; j += 256) s += ar[j] * tr[j] * sigma[j];
  sm[threadIdx.x] = s;
  __syncthreads();
  for (int ofs = 128; ofs > 0; ofs >>= 1) {
    if (threadIdx.x < ofs) sm[threadIdx.x] += sm[threadIdx.x + ofs];
    __syncthreads();
  }
  if (threadIdx.x == 0) delta[i] = (1.0f / n) / sm[0];
}

// sigma[j] = (1/n) / sum_i A[i,j]*T[i,j]*delta[i]   (block: 32 cols x 8 row-groups)
__global__ __launch_bounds__(256) void ipot_sigma(const float* __restrict__ A,
                                                  const float* __restrict__ T,
                                                  const float* __restrict__ delta,
                                                  float* __restrict__ sigma, int n) {
  __shared__ float sm[256];
  const int col = blockIdx.x * 32 + (threadIdx.x & 31);
  const int rg  = threadIdx.x >> 5;
  float s = 0.f;
  for (int i = rg; i < n; i += 8) {
    const size_t idx = (size_t)i * n + col;
    s += A[idx] * T[idx] * delta[i];
  }
  sm[threadIdx.x] = s;
  __syncthreads();
  for (int ofs = 128; ofs >= 32; ofs >>= 1) {
    if (threadIdx.x < ofs) sm[threadIdx.x] += sm[threadIdx.x + ofs];
    __syncthreads();
  }
  if (threadIdx.x < 32)
    sigma[blockIdx.x * 32 + threadIdx.x] = (1.0f / n) / sm[threadIdx.x];
}

// T[i,j] = delta[i] * A[i,j] * T[i,j] * sigma[j]
__global__ __launch_bounds__(256) void ipot_update(float* __restrict__ T,
                                                   const float* __restrict__ A,
                                                   const float* __restrict__ delta,
                                                   const float* __restrict__ sigma,
                                                   int n) {
  const size_t total = (size_t)n * n;
  for (size_t idx = ((size_t)blockIdx.x * 256 + threadIdx.x) * 4; idx < total;
       idx += (size_t)gridDim.x * 1024) {
    const int i = (int)(idx / n);
    const int j = (int)(idx % n);
    const float d = delta[i];
    const float4 a = *reinterpret_cast<const float4*>(A + idx);
    float4 t = *reinterpret_cast<float4*>(T + idx);
    t.x = d * a.x * t.x * sigma[j + 0];
    t.y = d * a.y * t.y * sigma[j + 1];
    t.z = d * a.z * t.z * sigma[j + 2];
    t.w = d * a.w * t.w * sigma[j + 3];
    *reinterpret_cast<float4*>(T + idx) = t;
  }
}

__global__ __launch_bounds__(256) void loss_partial(const float* __restrict__ C,
                                                    const float* __restrict__ T,
                                                    float* __restrict__ part,
                                                    size_t total) {
  __shared__ float sm[256];
  float s = 0.f;
  for (size_t idx = (size_t)blockIdx.x * 256 + threadIdx.x; idx < total;
       idx += (size_t)gridDim.x * 256)
    s += C[idx] * T[idx];
  sm[threadIdx.x] = s;
  __syncthreads();
  for (int ofs = 128; ofs > 0; ofs >>= 1) {
    if (threadIdx.x < ofs) sm[threadIdx.x] += sm[threadIdx.x + ofs];
    __syncthreads();
  }
  if (threadIdx.x == 0) part[blockIdx.x] = sm[0];
}

__global__ __launch_bounds__(256) void loss_final(const float* __restrict__ part,
                                                  int np, float* __restrict__ out) {
  __shared__ float sm[256];
  float s = 0.f;
  for (int i = threadIdx.x; i < np; i += 256) s += part[i];
  sm[threadIdx.x] = s;
  __syncthreads();
  for (int ofs = 128; ofs > 0; ofs >>= 1) {
    if (threadIdx.x < ofs) sm[threadIdx.x] += sm[threadIdx.x + ofs];
    __syncthreads();
  }
  if (threadIdx.x == 0) out[0] = sm[0];
}

// ---------------------------------------------------------------------------

extern "C" void kernel_launch(void* const* d_in, const int* in_sizes, int n_in,
                              void* d_out, int out_size, void* d_ws, size_t ws_size,
                              hipStream_t stream) {
  (void)in_sizes; (void)n_in; (void)out_size; (void)ws_size;
  const float* x     = (const float*)d_in[0];
  const float* xo    = (const float*)d_in[1];
  const float* W_enc = (const float*)d_in[2];
  const float* b_enc = (const float*)d_in[3];
  const float* W_dec = (const float*)d_in[4];
  const float* b_dec = (const float*)d_in[5];

  const int Bt = 2048, U = 20000, F = 800;

  float* out  = (float*)d_out;
  float* E    = out;                        // (Bt, F)
  float* pred = out + (size_t)Bt * F;       // (Bt, U)
  float* loss = pred + (size_t)Bt * U;      // scalar

  float* ws  = (float*)d_ws;
  float* Eo  = ws;                          // (Bt, F)
  float* En  = Eo  + (size_t)Bt * F;        // (Bt, F)
  float* Eon = En  + (size_t)Bt * F;        // (Bt, F)
  float* Cm  = Eon + (size_t)Bt * F;        // (Bt, Bt)
  float* Am  = Cm  + (size_t)Bt * Bt;       // (Bt, Bt)
  float* Tm  = Am  + (size_t)Bt * Bt;       // (Bt, Bt)
  float* sg  = Tm  + (size_t)Bt * Bt;       // (Bt)
  float* dl  = sg + Bt;                     // (Bt)
  float* pt  = dl + Bt;                     // (1024)

  const dim3 blk(256);

  // encoder GEMMs (K = 20000), sigmoid+bias fused
  gemm_nt<0><<<dim3((F + 63) / 64, Bt / 128), blk, 0, stream>>>(
      x, W_enc, b_enc, E, nullptr, Bt, F, U);
  gemm_nt<0><<<dim3((F + 63) / 64, Bt / 128), blk, 0, stream>>>(
      xo, W_enc, b_enc, Eo, nullptr, Bt, F, U);

  // decoder GEMM (K = 800), bias fused
  gemm_nt<1><<<dim3((U + 63) / 64, Bt / 128), blk, 0, stream>>>(
      E, W_dec, b_dec, pred, nullptr, Bt, U, F);

  // cosine cost: normalize rows then NT GEMM; fused C and A = exp(-C/beta)
  row_normalize<<<Bt, blk, 0, stream>>>(E, En, F);
  row_normalize<<<Bt, blk, 0, stream>>>(Eo, Eon, F);
  gemm_nt<2><<<dim3(Bt / 64, Bt / 128), blk, 0, stream>>>(
      En, Eon, nullptr, Cm, Am, Bt, Bt, F);

  // IPOT init: T = 1, sigma = 1/n
  fill_f32<<<4096, blk, 0, stream>>>(Tm, (size_t)Bt * Bt, 1.0f);
  fill_f32<<<8, blk, 0, stream>>>(sg, (size_t)Bt, 1.0f / Bt);

  for (int it = 0; it < 50; ++it) {
    ipot_delta<<<Bt, blk, 0, stream>>>(Am, Tm, sg, dl, Bt);
    ipot_sigma<<<Bt / 32, blk, 0, stream>>>(Am, Tm, dl, sg, Bt);
    ipot_update<<<4096, blk, 0, stream>>>(Tm, Am, dl, sg, Bt);
  }

  // ot_loss = sum(C * T), deterministic two-stage reduction
  loss_partial<<<1024, blk, 0, stream>>>(Cm, Tm, pt, (size_t)Bt * Bt);
  loss_final<<<1, blk, 0, stream>>>(pt, 1024, loss);
}
